// ImprovedModalGenerator_83262236000543
// MI455X (gfx1250) — matrix operand order repaired and checked by reading, not compiled, not run on hardware
//
#include <hip/hip_runtime.h>
#include <cmath>

#define VOCAB 49408
#define START_TOK 49406
#define END_TOK 49407
#define BATCH 128

typedef __attribute__((ext_vector_type(16))) _Float16 v16h;
typedef __attribute__((ext_vector_type(8)))  _Float16 v8h;
typedef __attribute__((ext_vector_type(8)))  float    v8f;
typedef __attribute__((ext_vector_type(4)))  float    v4f;

__device__ __forceinline__ float gelu_exact(float x) {
    return 0.5f * x * (1.0f + erff(x * 0.70710678118654752f));
}

__device__ __forceinline__ unsigned int pack2h(float a, float b) {
#if __has_builtin(__builtin_amdgcn_cvt_pkrtz)
    typedef __fp16 fp16x2 __attribute__((ext_vector_type(2)));
    fp16x2 v = __builtin_amdgcn_cvt_pkrtz(a, b);
    return __builtin_bit_cast(unsigned int, v);
#else
    union { unsigned int u; _Float16 h[2]; } p;
    p.h[0] = (_Float16)a; p.h[1] = (_Float16)b;
    return p.u;
#endif
}

// ---------------- avg pool 224x224 -> 14x14 (16x16 blocks), f16 out ----------------
__global__ void pool_kernel(const float* __restrict__ img, _Float16* __restrict__ pooled) {
    int o = blockIdx.x * blockDim.x + threadIdx.x;
    if (o >= BATCH * 588) return;
    int j = o % 14, i = (o / 14) % 14, c = (o / 196) % 3, b = o / 588;
    const float* base = img + (((size_t)b * 3 + c) * 224 + i * 16) * 224 + j * 16;
    float s = 0.f;
    for (int di = 0; di < 16; ++di) {
        const float* row = base + di * 224;
        #pragma unroll
        for (int dj = 0; dj < 16; ++dj) s += row[dj];
    }
    pooled[o] = (_Float16)(s * (1.0f / 256.0f));
}

// ---------------- embedding gather + mean over 77 tokens, f16 out ----------------
__global__ void emb_mean_kernel(const int* __restrict__ text, const float* __restrict__ emb,
                                _Float16* __restrict__ tfeat) {
    __shared__ int toks[77];
    int b = blockIdx.x;
    if (threadIdx.x < 77) toks[threadIdx.x] = text[b * 77 + threadIdx.x];
    __syncthreads();
    for (int d = threadIdx.x; d < 512; d += blockDim.x) {
        float s = 0.f;
        for (int t = 0; t < 77; ++t) s += emb[(size_t)toks[t] * 512 + d];
        tfeat[b * 512 + d] = (_Float16)(s * (1.0f / 77.0f));
    }
}

// ---------------- generated_mask = stack([miss==2, miss==1], axis=1) ----------------
__global__ void mask_kernel(const int* __restrict__ miss, float* __restrict__ out) {
    int t = threadIdx.x;
    if (t < 2 * BATCH) {
        int b = t >> 1, which = t & 1;
        int m = miss[b];
        out[t] = (which == 0) ? (m == 2 ? 1.f : 0.f) : (m == 1 ? 1.f : 0.f);
    }
}

// ---------------- WMMA GEMM: act(A[128,K](f16) @ W[K,N](f32) + bias) ----------------
// Block: 256 threads = 8 waves; block tile 128(M) x 128(N).
// Wave (wm=w&3, wn=w>>2): rows [wm*32,+32), cols [wn*64,+64). 8 wmma / K-step.
// LDS: pair-packed, double-buffered: sP[buf][kpair*132 + n] = {f16 W[k0+2kp][n], f16 W[k0+2kp+1][n]}
// act: 0=none(f16 out) 1=gelu(f16 out) 2=image tanh/clip/select(f32) 3=text token/select(f32)
template <bool GUARD>
__global__ __launch_bounds__(256)
void gemm_wmma(const _Float16* __restrict__ A, const float* __restrict__ W,
               const float* __restrict__ bias,
               _Float16* __restrict__ outh, float* __restrict__ outf,
               int K, int N, int act,
               const float* __restrict__ img, const int* __restrict__ texti,
               const int* __restrict__ miss) {
    __shared__ unsigned int sP[2][16 * 132];

    const int tid  = threadIdx.x;
    const int lane = tid & 31;
    const int wave = tid >> 5;
    const int wm   = wave & 3;
    const int wn   = wave >> 2;
    const int nb   = blockIdx.x * 128;
    const int n16  = lane & 15;
    const int kh   = lane >> 4;
    const int nK   = (K + 31) >> 5;

    v8f acc[2][4];
    #pragma unroll
    for (int s = 0; s < 2; ++s)
        #pragma unroll
        for (int t = 0; t < 4; ++t) {
            v8f z = {0.f, 0.f, 0.f, 0.f, 0.f, 0.f, 0.f, 0.f};
            acc[s][t] = z;
        }

    // W staging: thread owns k-pair row (2 global rows) x 8 columns
    const int fkp = tid >> 4;          // 0..15  (k = k0 + 2*fkp, +1)
    const int fn  = (tid & 15) * 8;    // 0..120

    alignas(16) float r0[8], r1[8];
    {   // prologue: stage kt = 0
        const int kA = 2 * fkp, kB = kA + 1;
        if (GUARD) {
            #pragma unroll
            for (int j = 0; j < 8; ++j) {
                int n = nb + fn + j;
                r0[j] = (kA < K && n < N) ? W[(size_t)kA * N + n] : 0.f;
                r1[j] = (kB < K && n < N) ? W[(size_t)kB * N + n] : 0.f;
            }
        } else {
            const float* w0 = W + (size_t)kA * N + nb + fn;
            *(v4f*)&r0[0] = *(const v4f*)w0;       *(v4f*)&r0[4] = *(const v4f*)(w0 + 4);
            *(v4f*)&r1[0] = *(const v4f*)(w0 + N); *(v4f*)&r1[4] = *(const v4f*)(w0 + N + 4);
        }
    }

    for (int kt = 0; kt < nK; ++kt) {
        const int k0  = kt * 32;
        const int buf = kt & 1;
        {   // commit staged tile to LDS (pair-packed f16)
            alignas(16) unsigned int p[8];
            #pragma unroll
            for (int j = 0; j < 8; ++j) p[j] = pack2h(r0[j], r1[j]);
            unsigned int* d = &sP[buf][fkp * 132 + fn];
            *(uint4*)d       = *(const uint4*)&p[0];
            *(uint4*)(d + 4) = *(const uint4*)&p[4];
        }
        __syncthreads();
        // prefetch next tile into registers (overlaps WMMA below)
        if (kt + 1 < nK) {
            const int kA = k0 + 32 + 2 * fkp, kB = kA + 1;
            if (GUARD) {
                #pragma unroll
                for (int j = 0; j < 8; ++j) {
                    int n = nb + fn + j;
                    r0[j] = (kA < K && n < N) ? W[(size_t)kA * N + n] : 0.f;
                    r1[j] = (kB < K && n < N) ? W[(size_t)kB * N + n] : 0.f;
                }
            } else {
                const float* w0 = W + (size_t)kA * N + nb + fn;
                *(v4f*)&r0[0] = *(const v4f*)w0;       *(v4f*)&r0[4] = *(const v4f*)(w0 + 4);
                *(v4f*)&r1[0] = *(const v4f*)(w0 + N); *(v4f*)&r1[4] = *(const v4f*)(w0 + N + 4);
            }
        }

        // A fragments (f16 in memory): ISA 16-bit A 16x32 layout.
        // lane M=n16, half kh; e<8: K=kh*8+e ; e>=8: K=16+kh*8+(e-8) -> two 16B runs
        union AF { v16h v; v8h p[2]; } af[2];
        #pragma unroll
        for (int s = 0; s < 2; ++s) {
            const int m = wm * 32 + s * 16 + n16;
            const _Float16* arow = A + (size_t)m * K + k0;
            if (GUARD) {
                #pragma unroll
                for (int e = 0; e < 16; ++e) {
                    int koff = (e < 8) ? (kh * 8 + e) : (16 + kh * 8 + (e - 8));
                    af[s].v[e] = (k0 + koff < K) ? arow[koff] : (_Float16)0.f;
                }
            } else {
                af[s].p[0] = *(const v8h*)(arow + kh * 8);
                af[s].p[1] = *(const v8h*)(arow + 16 + kh * 8);
            }
        }

        // B fragments: v16h VGPR e = pair (K=kh*8+2e,+1) for e<4, (K=16+kh*8+2(e-4),+1) for e>=4, column n
        // -> one dword from pair-packed LDS, conflict-free (stride 132)
        #pragma unroll
        for (int t = 0; t < 4; ++t) {
            const int n = wn * 64 + t * 16 + n16;
            union BF { v16h v; unsigned int u[8]; } bf;
            const unsigned int* sb = &sP[buf][kh * 4 * 132 + n];
            #pragma unroll
            for (int e = 0; e < 8; ++e) bf.u[e] = sb[(e < 4 ? e : e + 4) * 132];
            #pragma unroll
            for (int s = 0; s < 2; ++s)
                acc[s][t] = __builtin_amdgcn_wmma_f32_16x16x32_f16(
                    false, af[s].v, false, bf.v, (short)0, acc[s][t], false, false);
        }
        __syncthreads();
    }

    // epilogue: C/D layout: VGPR r, lane: N=n16, M = r + 8*kh
    #pragma unroll
    for (int s = 0; s < 2; ++s) {
        const int mrow = wm * 32 + s * 16 + kh * 8;
        #pragma unroll
        for (int t = 0; t < 4; ++t) {
            const int ncol = nb + wn * 64 + t * 16 + n16;
            if (GUARD && ncol >= N) continue;
            const float bv = bias[ncol];
            #pragma unroll
            for (int r = 0; r < 8; ++r) {
                const int m = mrow + r;
                const float x = acc[s][t][r] + bv;
                const size_t idx = (size_t)m * N + ncol;
                if (act == 0) {
                    outh[idx] = (_Float16)x;
                } else if (act == 1) {
                    outh[idx] = (_Float16)gelu_exact(x);
                } else if (act == 2) {
                    float y = tanhf(x);
                    y = fminf(2.5f, fmaxf(-2.5f, y));
                    outf[idx] = (miss[m] == 2) ? y : img[idx];
                } else { // act == 3: token generation + select
                    float y = tanhf(x);
                    float tokf;
                    if (ncol == 0)          tokf = (float)START_TOK;
                    else if (ncol == N - 1) tokf = (float)END_TOK;
                    else {
                        int tk = (int)((y + 1.0f) * 0.5f * (float)(VOCAB - 1));
                        tk = tk < 0 ? 0 : (tk > (VOCAB - 1) ? (VOCAB - 1) : tk);
                        tokf = (float)tk;
                    }
                    outf[idx] = (miss[m] == 1) ? tokf : (float)texti[idx];
                }
            }
        }
    }
}

extern "C" void kernel_launch(void* const* d_in, const int* in_sizes, int n_in,
                              void* d_out, int out_size, void* d_ws, size_t ws_size,
                              hipStream_t stream) {
    const float* image  = (const float*)d_in[0];
    const int*   text   = (const int*)  d_in[1];
    const int*   miss   = (const int*)  d_in[2];
    const float* enc_w1 = (const float*)d_in[3];
    const float* enc_b1 = (const float*)d_in[4];
    const float* enc_w2 = (const float*)d_in[5];
    const float* enc_b2 = (const float*)d_in[6];
    const float* i2t_w1 = (const float*)d_in[7];
    const float* i2t_b1 = (const float*)d_in[8];
    const float* i2t_w2 = (const float*)d_in[9];
    const float* i2t_b2 = (const float*)d_in[10];
    const float* i2t_w3 = (const float*)d_in[11];
    const float* i2t_b3 = (const float*)d_in[12];
    const float* emb    = (const float*)d_in[13];
    const float* t2i_w1 = (const float*)d_in[14];
    const float* t2i_b1 = (const float*)d_in[15];
    const float* t2i_w2 = (const float*)d_in[16];
    const float* t2i_b2 = (const float*)d_in[17];
    const float* t2i_w3 = (const float*)d_in[18];
    const float* t2i_b3 = (const float*)d_in[19];

    // f16 intermediates in workspace
    _Float16* hws    = (_Float16*)d_ws;
    _Float16* pooled = hws;                        // 128*588
    _Float16* x1     = pooled + BATCH * 588;       // 128*768
    _Float16* x2     = x1     + BATCH * 768;       // 128*768
    _Float16* h1     = x2     + BATCH * 768;       // 128*512
    _Float16* h2     = h1     + BATCH * 512;       // 128*256
    _Float16* tfeat  = h2     + BATCH * 256;       // 128*512
    _Float16* g1     = tfeat  + BATCH * 512;       // 128*1024
    _Float16* g2     = g1     + BATCH * 1024;      // 128*2048

    float* out_img  = (float*)d_out;                          // 128*150528
    float* out_text = out_img  + (size_t)BATCH * 150528;      // 128*77
    float* out_mask = out_text + (size_t)BATCH * 77;          // 128*2

    // ---- image -> text branch ----
    pool_kernel<<<(BATCH * 588 + 255) / 256, 256, 0, stream>>>(image, pooled);
    gemm_wmma<true ><<<(768 + 127) / 128, 256, 0, stream>>>(pooled, enc_w1, enc_b1, x1, nullptr, 588, 768, 1, nullptr, nullptr, nullptr);
    gemm_wmma<false><<< 768 / 128,          256, 0, stream>>>(x1, enc_w2, enc_b2, x2, nullptr, 768, 768, 0, nullptr, nullptr, nullptr);
    gemm_wmma<false><<< 512 / 128,          256, 0, stream>>>(x2, i2t_w1, i2t_b1, h1, nullptr, 768, 512, 1, nullptr, nullptr, nullptr);
    gemm_wmma<false><<< 256 / 128,          256, 0, stream>>>(h1, i2t_w2, i2t_b2, h2, nullptr, 512, 256, 1, nullptr, nullptr, nullptr);
    gemm_wmma<true ><<< 1,                  256, 0, stream>>>(h2, i2t_w3, i2t_b3, nullptr, out_text, 256, 77, 3, nullptr, text, miss);

    // ---- text -> image branch ----
    emb_mean_kernel<<<BATCH, 256, 0, stream>>>(text, emb, tfeat);
    gemm_wmma<false><<<1024 / 128,   256, 0, stream>>>(tfeat, t2i_w1, t2i_b1, g1, nullptr, 512, 1024, 1, nullptr, nullptr, nullptr);
    gemm_wmma<false><<<2048 / 128,   256, 0, stream>>>(g1, t2i_w2, t2i_b2, g2, nullptr, 1024, 2048, 1, nullptr, nullptr, nullptr);
    gemm_wmma<false><<<150528 / 128, 256, 0, stream>>>(g2, t2i_w3, t2i_b3, nullptr, out_img, 2048, 150528, 2, image, nullptr, miss);

    // ---- mask ----
    mask_kernel<<<1, 256, 0, stream>>>(miss, out_mask);
}
